// RecognitionModel_7352984011360
// MI455X (gfx1250) — matrix operand order repaired and checked by reading, not compiled, not run on hardware
//
#include <hip/hip_runtime.h>
#include <hip/hip_bf16.h>

// ---------------- problem constants ----------------
constexpr int Bn   = 4096;
constexpr int NE   = 64;
constexpr int DE   = 96;
constexpr int NA   = 8;
constexpr int NACT = 16;
constexpr int DH   = 256;
constexpr int NH   = 4;
constexpr int OD   = 64;
// HD = 64

// ---------------- WMMA types ----------------
typedef __attribute__((ext_vector_type(16))) __bf16 v16bf;
typedef __attribute__((ext_vector_type(8)))  float  v8f;

union Frag16 { v16bf v; uint4 q[2]; unsigned short s[16]; };
union Pk8    { uint4 q; unsigned short s[8]; };

__device__ __forceinline__ unsigned short f2bf(float f) {
    unsigned int u = __float_as_uint(f);
    u += 0x7fffu + ((u >> 16) & 1u);       // round-to-nearest-even
    return (unsigned short)(u >> 16);
}

// A fragment (16x32 bf16, row-major source, K contiguous):
// lane(ln,kh): a[0..7]=A[m][k0+kh*8 ..], a[8..15]=A[m][k0+16+kh*8 ..]
__device__ __forceinline__ v16bf load_frag_a(const unsigned short* base, int rowlen,
                                             int m, int k0, int kh) {
    Frag16 f;
    const unsigned short* p = base + m * rowlen + k0 + kh * 8;
    f.q[0] = *(const uint4*)p;
    f.q[1] = *(const uint4*)(p + 16);
    return f.v;
}

// B fragment (32x16 bf16) from transposed weight Wt[N][K]:
// lane(ln,kh): b[0..7]=Wt[n][k0+kh*8 ..], b[8..15]=Wt[n][k0+16+kh*8 ..]
__device__ __forceinline__ v16bf load_frag_b(const unsigned short* base, int rowlen,
                                             int n, int k0, int kh) {
    Frag16 f;
    const unsigned short* p = base + n * rowlen + k0 + kh * 8;
    f.q[0] = *(const uint4*)p;
    f.q[1] = *(const uint4*)(p + 16);
    return f.v;
}

template<int KSTEPS>
__device__ __forceinline__ v8f wmma_kloop(const unsigned short* A, int arl, int am, int ak0,
                                          const unsigned short* Bt, int brl, int bn, int bk0,
                                          int kh, v8f acc) {
#pragma unroll
    for (int ks = 0; ks < KSTEPS; ++ks) {
        v16bf a = load_frag_a(A, arl, am, ak0 + ks * 32, kh);
        v16bf b = load_frag_b(Bt, brl, bn, bk0 + ks * 32, kh);
        acc = __builtin_amdgcn_wmma_f32_16x16x32_bf16(false, a, false, b,
                                                      (short)0, acc, false, false);
    }
    return acc;
}

// ---------------- LDS layout (aliased regions) ----------------
struct alignas(16) SmemC1 {            // stages 0-1
    unsigned short ent[NE * DE];       // 12 KB
    float          ha[16 * DH];        // 16 KB  (rows 8..15 zero)
};
struct alignas(16) SmemC2 {            // stages 3..10
    float          logits[NH * NA * NE];   // [4][8][64]  8 KB
    unsigned short w[NH * 16 * NE];        // [4][16][64] 8 KB
    float          red[32][8];
    float          rowmax[32];
    float          rowsum[32];
    unsigned short att[16 * DH];           // 8 KB
    unsigned short hid[16 * DH];           // 8 KB
    unsigned short m1 [16 * DH];           // 8 KB
};
union SmemC { SmemC1 s1; SmemC2 s2; };
struct alignas(16) Smem {
    unsigned short x [NE * DH];            // 32 KB  relu(he+ha)
    unsigned short q [16 * DH];            // 8 KB
    unsigned short k [NE * DH];            // 32 KB
    unsigned short vT[DH * NE];            // 32 KB  v transposed [ch][key]
    float          hmul[16];               // entity-mask multiplier (rows>=8 -> 0)
    SmemC c;
};

// ---------------- weight convert/transpose pre-pass ----------------
__global__ void convert_transpose_kernel(const float* __restrict__ src,
                                         unsigned short* __restrict__ dst,
                                         int K, int N) {
    int idx = blockIdx.x * blockDim.x + threadIdx.x;
    if (idx < K * N) {
        int k = idx / N, n = idx - k * N;
        dst[n * K + k] = f2bf(src[idx]);   // dst is [N][K] bf16
    }
}

// ---------------- fused recognition model ----------------
__global__ __launch_bounds__(256) void fused_recog_kernel(
    const float* __restrict__ entities,      // [B,64,96]
    const int*   __restrict__ entity_mask,   // [B,64]
    const int*   __restrict__ obs_mask,      // [B,64,64]
    const float* __restrict__ acts,          // [B,8,16]
    const float* __restrict__ Wa,            // [16,256] fp32
    const float* __restrict__ be,  const float* __restrict__ bout,
    const float* __restrict__ bm1, const float* __restrict__ bm2,
    const float* __restrict__ bl1, const float* __restrict__ bl2,
    const unsigned short* __restrict__ Wet,   // [256][96]
    const unsigned short* __restrict__ Wint,  // [768][256]
    const unsigned short* __restrict__ Woutt, // [256][256]
    const unsigned short* __restrict__ Wm1t,  // [256][256]
    const unsigned short* __restrict__ Wm2t,  // [64][256]
    const unsigned short* __restrict__ Wl1t,  // [256][256]
    const unsigned short* __restrict__ Wl2t,  // [64][256]
    float* __restrict__ out)                  // mean[B,8,64] ++ logvar[B,8,64]
{
    extern __shared__ char smem_raw[];
    Smem& sm = *reinterpret_cast<Smem*>(smem_raw);

    const int b    = blockIdx.x;
    const int tid  = threadIdx.x;
    const int wv   = tid >> 5;
    const int lane = tid & 31;
    const int ln   = lane & 15;
    const int kh   = lane >> 4;
    const float NEG_INF = -__builtin_inff();

    // ---- stage 0: entities -> LDS bf16; ha = onehot @ Wa (16 rows, 8..15 zero);
    //               hmul = entity-mask multiplier ----
    const float* entB = entities + (size_t)b * NE * DE;
    for (int i = tid; i < NE * DE; i += 256) sm.c.s1.ent[i] = f2bf(entB[i]);
    const float* actB = acts + (size_t)b * NA * NACT;
    for (int i = tid; i < 16 * DH; i += 256) {
        int qq = i >> 8, col = i & 255;
        float s = 0.f;
        if (qq < NA) {
#pragma unroll
            for (int a = 0; a < NACT; ++a) s += actB[qq * NACT + a] * Wa[a * DH + col];
        }
        sm.c.s1.ha[i] = s;
    }
    if (tid < 16)
        sm.hmul[tid] = (tid < NA) ? (entity_mask[(size_t)b * NE + tid] ? 0.f : 1.f) : 0.f;
    __syncthreads();

    // ---- stage 1a: x rows 0..15 = relu(ent @ We + be + ha) (unconditional ha add) ----
    for (int tt = wv; tt < 16; tt += 8) {
        int nt = tt;                                   // mt == 0
        v8f acc = {};
        acc = wmma_kloop<3>(sm.c.s1.ent, DE, ln, 0, Wet, DE, nt * 16 + ln, 0, kh, acc);
        int col = nt * 16 + ln;
        float bias = be[col];
#pragma unroll
        for (int j = 0; j < 8; ++j) {
            int row = kh * 8 + j;
            float v = acc[j] + bias + sm.c.s1.ha[row * DH + col];
            sm.x[row * DH + col] = f2bf(fmaxf(v, 0.f));
        }
    }
    // ---- stage 1b: x rows 16..63 = relu(ent @ We + be) ----
    for (int tt = 16 + wv; tt < 64; tt += 8) {
        int mt = tt >> 4, nt = tt & 15;
        v8f acc = {};
        acc = wmma_kloop<3>(sm.c.s1.ent, DE, mt * 16 + ln, 0, Wet, DE, nt * 16 + ln, 0, kh, acc);
        int col = nt * 16 + ln;
        float bias = be[col];
#pragma unroll
        for (int j = 0; j < 8; ++j) {
            int row = mt * 16 + kh * 8 + j;
            sm.x[row * DH + col] = f2bf(fmaxf(acc[j] + bias, 0.f));
        }
    }
    __syncthreads();

    // ---- stage 2: qkv = x @ Win (q rows 0..15; k [64,256]; vT [256,64]) ----
    for (int tt = wv; tt < 144; tt += 8) {
        int mt, colBase, mode;
        if (tt < 16)      { mt = 0;               colBase = tt * 16;              mode = 0; }
        else if (tt < 80) { int u = tt - 16; mt = u & 3; colBase = 256 + (u >> 2) * 16; mode = 1; }
        else              { int u = tt - 80; mt = u & 3; colBase = 512 + (u >> 2) * 16; mode = 2; }
        v8f acc = {};
        acc = wmma_kloop<8>(sm.x, DH, mt * 16 + ln, 0, Wint, DH, colBase + ln, 0, kh, acc);
        if (mode == 0) {
            int col = colBase + ln;
#pragma unroll
            for (int j = 0; j < 8; ++j) sm.q[(kh * 8 + j) * DH + col] = f2bf(acc[j]);
        } else if (mode == 1) {
            int col = colBase - 256 + ln;
#pragma unroll
            for (int j = 0; j < 8; ++j) sm.k[(mt * 16 + kh * 8 + j) * DH + col] = f2bf(acc[j]);
        } else {
            int ch = colBase - 512 + ln;            // channel 0..255
            Pk8 pk;
#pragma unroll
            for (int j = 0; j < 8; ++j) pk.s[j] = f2bf(acc[j]);
            *(uint4*)&sm.vT[ch * NE + mt * 16 + kh * 8] = pk.q;   // keys contiguous
        }
    }
    __syncthreads();

    // ---- stage 3: logits = q.kT / 8  -> [4][8][64] f32 ----
    for (int tt = wv; tt < 16; tt += 8) {
        int h = tt >> 2, nt = tt & 3;
        v8f acc = {};
        acc = wmma_kloop<2>(sm.q, DH, ln, h * 64, sm.k, DH, nt * 16 + ln, h * 64, kh, acc);
        if (kh == 0) {
            int key = nt * 16 + ln;
#pragma unroll
            for (int j = 0; j < 8; ++j)
                sm.c.s2.logits[(h * NA + j) * NE + key] = 0.125f * acc[j];
        }
    }
    __syncthreads();

    // ---- stage 4: mask + softmax (fully-masked rows -> 0) ----
    {
        int r = tid >> 3, i = tid & 7;          // r: 0..31 = h*8+q, i: 8-col chunks
        int h = r >> 3, qq = r & 7;
        const int* om = obs_mask + (size_t)b * NE * NE + qq * NE;
        float lv[8]; float mx = NEG_INF;
#pragma unroll
        for (int e = 0; e < 8; ++e) {
            int c = i * 8 + e;
            float v = om[c] ? NEG_INF : sm.c.s2.logits[r * NE + c];
            lv[e] = v; mx = fmaxf(mx, v);
        }
        sm.c.s2.red[r][i] = mx;
        __syncthreads();
        if (i == 0) {
            float m = sm.c.s2.red[r][0];
#pragma unroll
            for (int t = 1; t < 8; ++t) m = fmaxf(m, sm.c.s2.red[r][t]);
            sm.c.s2.rowmax[r] = m;
        }
        __syncthreads();
        mx = sm.c.s2.rowmax[r];
        float ev[8]; float s = 0.f;
#pragma unroll
        for (int e = 0; e < 8; ++e) {
            float v = (lv[e] == NEG_INF) ? 0.f : __expf(lv[e] - mx);
            ev[e] = v; s += v;
        }
        sm.c.s2.red[r][i] = s;
        __syncthreads();
        if (i == 0) {
            float m = 0.f;
#pragma unroll
            for (int t = 0; t < 8; ++t) m += sm.c.s2.red[r][t];
            sm.c.s2.rowsum[r] = m;
        }
        __syncthreads();
        float inv = sm.c.s2.rowsum[r] > 0.f ? 1.f / sm.c.s2.rowsum[r] : 0.f;
#pragma unroll
        for (int e = 0; e < 8; ++e)
            sm.c.s2.w[h * 1024 + qq * 64 + i * 8 + e] = f2bf(ev[e] * inv);
    }
    __syncthreads();

    // ---- stage 5: attn = w @ v  -> att [16,256] bf16 ----
    for (int tt = wv; tt < 16; tt += 8) {
        int h = tt >> 2, nt = tt & 3;
        v8f acc = {};
        acc = wmma_kloop<2>(sm.c.s2.w + h * 1024, 64, ln, 0,
                            sm.vT, NE, h * 64 + nt * 16 + ln, 0, kh, acc);
        int col = h * 64 + nt * 16 + ln;
#pragma unroll
        for (int j = 0; j < 8; ++j) sm.c.s2.att[(kh * 8 + j) * DH + col] = f2bf(acc[j]);
    }
    __syncthreads();

    // ---- stage 6: hidden = relu(att @ Wout + bout) * hmul (branch-free mask) ----
    for (int nt = wv; nt < 16; nt += 8) {
        v8f acc = {};
        acc = wmma_kloop<8>(sm.c.s2.att, DH, ln, 0, Woutt, DH, nt * 16 + ln, 0, kh, acc);
        int col = nt * 16 + ln;
        float bias = bout[col];
#pragma unroll
        for (int j = 0; j < 8; ++j) {
            int row = kh * 8 + j;
            float hv = fmaxf(acc[j] + bias, 0.f) * sm.hmul[row];
            sm.c.s2.hid[row * DH + col] = f2bf(hv);
        }
    }
    __syncthreads();

    // ---- stage 7: m1 = relu(hidden @ Wm1 + bm1) ----
    for (int nt = wv; nt < 16; nt += 8) {
        v8f acc = {};
        acc = wmma_kloop<8>(sm.c.s2.hid, DH, ln, 0, Wm1t, DH, nt * 16 + ln, 0, kh, acc);
        int col = nt * 16 + ln;
        float bias = bm1[col];
#pragma unroll
        for (int j = 0; j < 8; ++j)
            sm.c.s2.m1[(kh * 8 + j) * DH + col] = f2bf(fmaxf(acc[j] + bias, 0.f));
    }
    __syncthreads();

    // ---- stage 8: mean = m1 @ Wm2 + bm2 -> out ----
    if (wv < 4) {
        int nt = wv;
        v8f acc = {};
        acc = wmma_kloop<8>(sm.c.s2.m1, DH, ln, 0, Wm2t, DH, nt * 16 + ln, 0, kh, acc);
        int col = nt * 16 + ln;
        float bias = bm2[col];
        if (kh == 0) {
#pragma unroll
            for (int j = 0; j < 8; ++j)
                out[((size_t)b * NA + j) * OD + col] = acc[j] + bias;
        }
    }
    __syncthreads();

    // ---- stage 9: l1 = relu(hidden @ Wl1 + bl1)  (reuse m1 buffer) ----
    for (int nt = wv; nt < 16; nt += 8) {
        v8f acc = {};
        acc = wmma_kloop<8>(sm.c.s2.hid, DH, ln, 0, Wl1t, DH, nt * 16 + ln, 0, kh, acc);
        int col = nt * 16 + ln;
        float bias = bl1[col];
#pragma unroll
        for (int j = 0; j < 8; ++j)
            sm.c.s2.m1[(kh * 8 + j) * DH + col] = f2bf(fmaxf(acc[j] + bias, 0.f));
    }
    __syncthreads();

    // ---- stage 10: logvar = clip(l1 @ Wl2 + bl2, -10, 0) -> out ----
    if (wv < 4) {
        int nt = wv;
        v8f acc = {};
        acc = wmma_kloop<8>(sm.c.s2.m1, DH, ln, 0, Wl2t, DH, nt * 16 + ln, 0, kh, acc);
        int col = nt * 16 + ln;
        float bias = bl2[col];
        if (kh == 0) {
            float* lout = out + (size_t)Bn * NA * OD;
#pragma unroll
            for (int j = 0; j < 8; ++j) {
                float v = acc[j] + bias;
                lout[((size_t)b * NA + j) * OD + col] = fminf(fmaxf(v, -10.f), 0.f);
            }
        }
    }
}

// ---------------- host launch ----------------
extern "C" void kernel_launch(void* const* d_in, const int* in_sizes, int n_in,
                              void* d_out, int out_size, void* d_ws, size_t ws_size,
                              hipStream_t stream) {
    const float* entities    = (const float*)d_in[0];
    const int*   entity_mask = (const int*)  d_in[1];
    const int*   obs_mask    = (const int*)  d_in[2];
    const float* acts        = (const float*)d_in[3];
    const float* We   = (const float*)d_in[4];
    const float* be   = (const float*)d_in[5];
    const float* Wa   = (const float*)d_in[6];
    const float* Win  = (const float*)d_in[7];
    const float* Wout = (const float*)d_in[8];
    const float* bout = (const float*)d_in[9];
    const float* Wm1  = (const float*)d_in[10];
    const float* bm1  = (const float*)d_in[11];
    const float* Wm2  = (const float*)d_in[12];
    const float* bm2  = (const float*)d_in[13];
    const float* Wl1  = (const float*)d_in[14];
    const float* bl1  = (const float*)d_in[15];
    const float* Wl2  = (const float*)d_in[16];
    const float* bl2  = (const float*)d_in[17];
    float* out = (float*)d_out;

    // bf16 transposed weights in workspace (~0.9 MB)
    unsigned short* ws    = (unsigned short*)d_ws;
    unsigned short* Wet   = ws;                       // [256][96]
    unsigned short* Wint  = Wet  + 96 * 256;          // [768][256]
    unsigned short* Woutt = Wint + 256 * 768;         // [256][256]
    unsigned short* Wm1t  = Woutt + 256 * 256;        // [256][256]
    unsigned short* Wm2t  = Wm1t + 256 * 256;         // [64][256]
    unsigned short* Wl1t  = Wm2t + 256 * 64;          // [256][256]
    unsigned short* Wl2t  = Wl1t + 256 * 256;         // [64][256]

    auto T = [&](const float* src, unsigned short* dst, int K, int N) {
        int total = K * N;
        convert_transpose_kernel<<<(total + 255) / 256, 256, 0, stream>>>(src, dst, K, N);
    };
    T(We,   Wet,   96, 256);
    T(Win,  Wint,  256, 768);
    T(Wout, Woutt, 256, 256);
    T(Wm1,  Wm1t,  256, 256);
    T(Wm2,  Wm2t,  256, 64);
    T(Wl1,  Wl1t,  256, 256);
    T(Wl2,  Wl2t,  256, 64);

    size_t shmem = sizeof(Smem);   // ~145 KB dynamic LDS
    fused_recog_kernel<<<dim3(Bn), dim3(256), shmem, stream>>>(
        entities, entity_mask, obs_mask, acts, Wa,
        be, bout, bm1, bm2, bl1, bl2,
        Wet, Wint, Woutt, Wm1t, Wm2t, Wl1t, Wl2t, out);
}
